// BitLinear_35605278884137
// MI455X (gfx1250) — 8-bit matrix operand order repaired and checked by reading, compiled, not run on hardware
//
#include <hip/hip_runtime.h>
#include <stdint.h>

typedef int v8i __attribute__((ext_vector_type(8)));

#define D_K   2048
#define D_N   2048
#define NROWS 16384

#define XQ_OFF    ((size_t)0)
#define WQ_OFF    ((size_t)(NROWS) * D_K)
#define RINV_OFF  (WQ_OFF + (size_t)D_N * D_K)
#define PART_OFF  (RINV_OFF + (size_t)NROWS * 4)
#define SCALE_OFF (PART_OFF + (size_t)D_N * 4)

__global__ void k_init(float* scales) {
    if (threadIdx.x < 8) scales[threadIdx.x] = 0.0f;
}

__global__ __launch_bounds__(256)
void k_row_stats(const float* __restrict__ x, const float* __restrict__ gamma,
                 float* __restrict__ rowinv, float* __restrict__ scales) {
    __shared__ float s_sum[256];
    __shared__ float s_max[256];
    const int row = blockIdx.x;
    const float* xr = x + (size_t)row * D_K;
    float ss = 0.0f, mx = 0.0f;
    for (int i = threadIdx.x; i < D_K; i += 256) {
        float v = xr[i];
        ss += v * v;
        mx = fmaxf(mx, fabsf(v * gamma[i]));
    }
    s_sum[threadIdx.x] = ss;
    s_max[threadIdx.x] = mx;
    __syncthreads();
    for (int s = 128; s > 0; s >>= 1) {
        if (threadIdx.x < s) {
            s_sum[threadIdx.x] += s_sum[threadIdx.x + s];
            s_max[threadIdx.x] = fmaxf(s_max[threadIdx.x], s_max[threadIdx.x + s]);
        }
        __syncthreads();
    }
    if (threadIdx.x == 0) {
        float mean = s_sum[0] * (1.0f / (float)D_K);
        float inv  = rsqrtf(mean + 1e-6f);
        rowinv[row] = inv;
        float rowmax = s_max[0] * inv;
        atomicMax((unsigned int*)&scales[0], __float_as_uint(rowmax));
    }
}

__global__ __launch_bounds__(256)
void k_w_abs(const float* __restrict__ w, float* __restrict__ partial) {
    __shared__ float s[256];
    const int row = blockIdx.x;
    const float* wr = w + (size_t)row * D_K;
    float ss = 0.0f;
    for (int i = threadIdx.x; i < D_K; i += 256) ss += fabsf(wr[i]);
    s[threadIdx.x] = ss;
    __syncthreads();
    for (int st = 128; st > 0; st >>= 1) {
        if (threadIdx.x < st) s[threadIdx.x] += s[threadIdx.x + st];
        __syncthreads();
    }
    if (threadIdx.x == 0) partial[row] = s[0];
}

__global__ __launch_bounds__(256)
void k_finalize(const float* __restrict__ partial, float* __restrict__ scales) {
    __shared__ float s[256];
    float ss = 0.0f;
    for (int i = threadIdx.x; i < D_N; i += 256) ss += partial[i];
    s[threadIdx.x] = ss;
    __syncthreads();
    for (int st = 128; st > 0; st >>= 1) {
        if (threadIdx.x < st) s[threadIdx.x] += s[threadIdx.x + st];
        __syncthreads();
    }
    if (threadIdx.x == 0) {
        float wmean = s[0] * (1.0f / ((float)D_N * (float)D_K));
        float wsc = fmaxf(wmean, 1e-5f);
        float asc = fmaxf(scales[0], 1e-5f);
        scales[1] = wsc;
        scales[2] = asc;
        scales[3] = asc * wsc * (1.0f / 127.0f);
    }
}

__global__ __launch_bounds__(256)
void k_quant_act(const float* __restrict__ x, const float* __restrict__ gamma,
                 const float* __restrict__ rowinv, const float* __restrict__ scales,
                 uint32_t* __restrict__ xq) {
    const size_t gid = (size_t)blockIdx.x * 256 + threadIdx.x;
    const size_t idx = gid * 4;
    const int row = (int)(idx >> 11);
    const int col = (int)(idx & 2047);
    const float f = rowinv[row] * (127.0f / scales[2]);
    uint32_t pack = 0;
#pragma unroll
    for (int j = 0; j < 4; ++j) {
        float v = x[idx + j] * gamma[col + j] * f;
        v = fminf(fmaxf(v, -128.0f), 127.0f);
        int q = (int)rintf(v);
        pack |= ((uint32_t)(uint8_t)(int8_t)q) << (8 * j);
    }
    xq[gid] = pack;
}

__global__ __launch_bounds__(256)
void k_quant_w(const float* __restrict__ w, const float* __restrict__ scales,
               uint32_t* __restrict__ wq) {
    const size_t gid = (size_t)blockIdx.x * 256 + threadIdx.x;
    const size_t idx = gid * 4;
    const float invws = 1.0f / scales[1];
    uint32_t pack = 0;
#pragma unroll
    for (int j = 0; j < 4; ++j) {
        float v = w[idx + j] * invws;
        v = fminf(fmaxf(v, -1.0f), 1.0f);
        int q = (int)rintf(v);
        pack |= ((uint32_t)(uint8_t)(int8_t)q) << (8 * j);
    }
    wq[gid] = pack;
}

__device__ __forceinline__ v8i load_a_frag(const uint32_t* p) {
    v8i v;
    v[0] = p[0];  v[1] = p[1];  v[2] = p[4];  v[3] = p[5];
    v[4] = p[8];  v[5] = p[9];  v[6] = p[12]; v[7] = p[13];
    return v;
}
__device__ __forceinline__ v8i load_b_frag(const uint32_t* p) {
    v8i v;
    v[0] = p[0];  v[1] = p[1];  v[2] = p[4];  v[3] = p[5];
    v[4] = p[8];  v[5] = p[9];  v[6] = p[12]; v[7] = p[13];
    return v;
}

#define WMMA_IU8(a, b, c) \
    __builtin_amdgcn_wmma_i32_16x16x64_iu8(true, (a), true, (b), (c), false, false)

__global__ __launch_bounds__(256)
void k_gemm_iu8(const int8_t* __restrict__ Xq, const int8_t* __restrict__ Wq,
                const float* __restrict__ scales, float* __restrict__ out) {
    const int lane = threadIdx.x & 31;
    const int wave = threadIdx.x >> 5;
    const int wm = wave >> 2;
    const int wn = wave & 3;
    const int m0 = blockIdx.x * 128 + wm * 64;
    const int n0 = blockIdx.y * 128 + wn * 32;
    const int lrow = lane & 15;
    const int ka = (lane >> 4) * 8;
    const int kb = (lane >> 4) * 8;

    const uint32_t* A0 = (const uint32_t*)(Xq + (size_t)(m0 + lrow)      * D_K + ka);
    const uint32_t* A1 = (const uint32_t*)(Xq + (size_t)(m0 + 16 + lrow) * D_K + ka);
    const uint32_t* A2 = (const uint32_t*)(Xq + (size_t)(m0 + 32 + lrow) * D_K + ka);
    const uint32_t* A3 = (const uint32_t*)(Xq + (size_t)(m0 + 48 + lrow) * D_K + ka);
    const uint32_t* B0 = (const uint32_t*)(Wq + (size_t)(n0 + lrow)      * D_K + kb);
    const uint32_t* B1 = (const uint32_t*)(Wq + (size_t)(n0 + 16 + lrow) * D_K + kb);

    v8i c00 = {}, c01 = {};
    v8i c10 = {}, c11 = {};
    v8i c20 = {}, c21 = {};
    v8i c30 = {}, c31 = {};

    for (int kk = 0; kk < D_K; kk += 64) {
        const int dw = kk >> 2;
        __builtin_prefetch((const char*)(A0 + dw) + 64, 0, 1);
        __builtin_prefetch((const char*)(B0 + dw) + 64, 0, 1);

        v8i b0 = load_b_frag(B0 + dw);
        v8i b1 = load_b_frag(B1 + dw);
        v8i a0 = load_a_frag(A0 + dw);
        v8i a1 = load_a_frag(A1 + dw);
        v8i a2 = load_a_frag(A2 + dw);
        v8i a3 = load_a_frag(A3 + dw);

        c00 = WMMA_IU8(a0, b0, c00);
        c01 = WMMA_IU8(a0, b1, c01);
        c10 = WMMA_IU8(a1, b0, c10);
        c11 = WMMA_IU8(a1, b1, c11);
        c20 = WMMA_IU8(a2, b0, c20);
        c21 = WMMA_IU8(a2, b1, c21);
        c30 = WMMA_IU8(a3, b0, c30);
        c31 = WMMA_IU8(a3, b1, c31);
    }

    const float sc = scales[3];
    const int orow = m0 + (lane >> 4) * 8;
    const int ocol = n0 + (lane & 15);
#pragma unroll
    for (int r = 0; r < 8; ++r) {
        out[(size_t)(orow + r)      * D_N + ocol]      = (float)c00[r] * sc;
        out[(size_t)(orow + r)      * D_N + ocol + 16] = (float)c01[r] * sc;
        out[(size_t)(orow + r + 16) * D_N + ocol]      = (float)c10[r] * sc;
        out[(size_t)(orow + r + 16) * D_N + ocol + 16] = (float)c11[r] * sc;
        out[(size_t)(orow + r + 32) * D_N + ocol]      = (float)c20[r] * sc;
        out[(size_t)(orow + r + 32) * D_N + ocol + 16] = (float)c21[r] * sc;
        out[(size_t)(orow + r + 48) * D_N + ocol]      = (float)c30[r] * sc;
        out[(size_t)(orow + r + 48) * D_N + ocol + 16] = (float)c31[r] * sc;
    }
}

extern "C" void kernel_launch(void* const* d_in, const int* in_sizes, int n_in,
                              void* d_out, int out_size, void* d_ws, size_t ws_size,
                              hipStream_t stream) {
    const float* x     = (const float*)d_in[0];
    const float* w     = (const float*)d_in[1];
    const float* gamma = (const float*)d_in[2];
    float* out = (float*)d_out;

    int8_t*   Xq     = (int8_t*)((char*)d_ws + XQ_OFF);
    int8_t*   Wq     = (int8_t*)((char*)d_ws + WQ_OFF);
    float*    rowinv = (float*)((char*)d_ws + RINV_OFF);
    float*    part   = (float*)((char*)d_ws + PART_OFF);
    float*    scales = (float*)((char*)d_ws + SCALE_OFF);

    k_init<<<1, 32, 0, stream>>>(scales);
    k_row_stats<<<NROWS, 256, 0, stream>>>(x, gamma, rowinv, scales);
    k_w_abs<<<D_N, 256, 0, stream>>>(w, part);
    k_finalize<<<1, 256, 0, stream>>>(part, scales);
    k_quant_act<<<(NROWS * (D_K / 4)) / 256, 256, 0, stream>>>(x, gamma, rowinv, scales,
                                                              (uint32_t*)Xq);
    k_quant_w<<<(D_N * (D_K / 4)) / 256, 256, 0, stream>>>(w, scales, (uint32_t*)Wq);

    dim3 grid(NROWS / 128, D_N / 128);
    k_gemm_iu8<<<grid, 256, 0, stream>>>(Xq, Wq, scales, out);
}
